// SelfAttentionLayer_5772436046133
// MI455X (gfx1250) — matrix operand order repaired and checked by reading, not compiled, not run on hardware
//
#include <hip/hip_runtime.h>
#include <hip/hip_bf16.h>

// ---------------------------------------------------------------------------
// Fused row-attention for MI455X (gfx1250, wave32, bf16 WMMA 16x16x32).
// B=16, H=128, W=128, C=256. One workgroup (8 waves) per (b,h) row.
// LDS (dynamic, 266 KiB of the WGP's 320 KiB):
//   [0      , 67584 )  xb : x row as bf16 [128][264]   (reused as s f32 [128][132])
//   [67584  , 135168)  fB : f bf16 [128][264]          (reused as beta bf16 [128][136])
//   [135168 , 202752)  gB : g bf16 [128][264]
//   [202752 , 272384)  hT : h^T bf16 [256][136]
// Weights pre-transposed to bf16 in workspace (L2-resident, 384 KiB).
// Register-pressure design: no loop-invariant fragment loads anywhere
// (prevents LICM from caching 64 fragments and spilling to scratch).
// ---------------------------------------------------------------------------

typedef __bf16 bf16_t;
typedef __attribute__((ext_vector_type(16))) __bf16 v16bf;
typedef __attribute__((ext_vector_type(8)))  __bf16 v8bf;
typedef __attribute__((ext_vector_type(8)))  float  v8f;

#define SA_W 128
#define SA_C 256
#define PITCH_X  264   // halfs (256 + 8 pad)
#define PITCH_S  132   // floats (128 + 4 pad)
#define PITCH_B  136   // halfs (128 + 8 pad) for beta and hT
#define OFF_F   67584
#define OFF_G  135168
#define OFF_HT 202752
#define SMEM_BYTES 272384

static __device__ inline v8f zero8() {
  v8f z = {0.f, 0.f, 0.f, 0.f, 0.f, 0.f, 0.f, 0.f};
  return z;
}

// Load a 16x32 bf16 A-operand fragment from a row-major array.
// p points at element (R0, K0); pitch in halfs.
// Lane l in 0-15: row R0+l, K = {0..7, 16..23}; lane 16+l: K = {8..15, 24..31}.
static __device__ inline v16bf load_frag_a(const bf16_t* p, int pitch) {
  const int l = threadIdx.x & 31;
  const bf16_t* q = p + (l & 15) * pitch + ((l >> 4) << 3);
  v8bf lo = *(const v8bf*)(q);
  v8bf hi = *(const v8bf*)(q + 16);
  return __builtin_shufflevector(lo, hi, 0, 1, 2, 3, 4, 5, 6, 7,
                                 8, 9, 10, 11, 12, 13, 14, 15);
}

// Load a 32x16 bf16 B-operand fragment. Columns of B are rows of the given
// row-major array (i.e. pass B^T). Lane l in 0-15: col N0+l, K = {0..7, 16..23};
// lane 16+l: col N0+l, K = {8..15, 24..31}.
static __device__ inline v16bf load_frag_b(const bf16_t* p, int pitch) {
  const int l = threadIdx.x & 31;
  const bf16_t* q = p + (l & 15) * pitch + ((l >> 4) << 3);
  v8bf lo = *(const v8bf*)(q);
  v8bf hi = *(const v8bf*)(q + 16);
  return __builtin_shufflevector(lo, hi, 0, 1, 2, 3, 4, 5, 6, 7,
                                 8, 9, 10, 11, 12, 13, 14, 15);
}

static __device__ inline v8f wmma_bf16(v16bf a, v16bf b, v8f c) {
  // D = A(16x32) * B(32x16) + C, f32 accumulate.
  return __builtin_amdgcn_wmma_f32_16x16x32_bf16(
      /*neg_a=*/false, a, /*neg_b=*/false, b,
      /*c_mod=*/(short)0, c, /*reuse_a=*/false, /*reuse_b=*/false);
}

// Pre-transpose + convert weights: wt[p][n][k] = W_p[k][n] as bf16.
__global__ void sa_prep_weights(const float* __restrict__ Wf,
                                const float* __restrict__ Wg,
                                const float* __restrict__ Wh,
                                bf16_t* __restrict__ wt) {
  int idx = blockIdx.x * blockDim.x + threadIdx.x;
  if (idx >= 3 * SA_C * SA_C) return;
  int p = idx >> 16;
  int r = idx & 0xFFFF;
  int n = r >> 8;
  int k = r & 255;
  const float* src = (p == 0) ? Wf : ((p == 1) ? Wg : Wh);
  wt[idx] = (bf16_t)src[k * SA_C + n];
}

__global__ void __launch_bounds__(256)
sa_row_attention(const float* __restrict__ x,
                 const bf16_t* __restrict__ wt,
                 const float* __restrict__ bfv,
                 const float* __restrict__ bgv,
                 const float* __restrict__ bhv,
                 float* __restrict__ out) {
  extern __shared__ char smem[];
  bf16_t* xb   = (bf16_t*)smem;            // phase 1-2
  float*  sS   = (float*)smem;             // phase 3+ (same space)
  bf16_t* fB   = (bf16_t*)(smem + OFF_F);
  bf16_t* gB   = (bf16_t*)(smem + OFF_G);
  bf16_t* hT   = (bf16_t*)(smem + OFF_HT);
  bf16_t* beta = fB;                       // reuse f's space, pitch PITCH_B

  const int bid  = blockIdx.x;             // b*H + h, 0..2047
  const float* xrow = x + (size_t)bid * (SA_W * SA_C);
  float*       orow = out + (size_t)bid * (SA_W * SA_C);
  const int tid  = threadIdx.x;
  const int lane = tid & 31;
  const int wave = tid >> 5;               // 0..7

  // ---- Phase 1: x row (fp32) -> bf16 in LDS, coalesced float4 loads ----
  for (int i = tid; i < (SA_W * SA_C) / 4; i += 256) {
    int e = i * 4;
    int w = e >> 8;
    int c = e & 255;
    float4 v = *(const float4*)(xrow + e);
    bf16_t* d = xb + w * PITCH_X + c;
    d[0] = (bf16_t)v.x; d[1] = (bf16_t)v.y;
    d[2] = (bf16_t)v.z; d[3] = (bf16_t)v.w;
  }
  __syncthreads();

  // ---- Phase 2: f,g,h projections. Wave grid: 4 (M) x 2 (N halves). ----
  // Dual-M accumulation: every B-fragment load feeds two WMMAs immediately;
  // no fragment load is invariant w.r.t. any enclosing loop.
  const int wm  = wave >> 1;               // M-tile pair {2*wm, 2*wm+1}
  const int wn  = wave & 1;                // N-tiles {8*wn .. 8*wn+7}
  const int m0a = (2 * wm) * 16;
  const int m0b = m0a + 16;
#pragma clang loop unroll(disable)
  for (int p = 0; p < 3; ++p) {
    const bf16_t* wp = wt + p * (SA_C * SA_C);
    const float* bias = (p == 0) ? bfv : ((p == 1) ? bgv : bhv);
    v8f acc0[8], acc1[8];
#pragma unroll
    for (int n = 0; n < 8; ++n) { acc0[n] = zero8(); acc1[n] = zero8(); }
#pragma clang loop unroll(disable)
    for (int k = 0; k < 8; ++k) {
      const int k0 = k * 32;
      v16bf a0 = load_frag_a(xb + m0a * PITCH_X + k0, PITCH_X);
      v16bf a1 = load_frag_a(xb + m0b * PITCH_X + k0, PITCH_X);
#pragma unroll
      for (int n = 0; n < 8; ++n) {
        const int n0 = (wn * 8 + n) * 16;
        v16bf b = load_frag_b(wp + n0 * SA_C + k0, SA_C);
        acc0[n] = wmma_bf16(a0, b, acc0[n]);
        acc1[n] = wmma_bf16(a1, b, acc1[n]);
      }
    }
    const int rb0 = m0a + ((lane >> 4) << 3);
    const int rb1 = m0b + ((lane >> 4) << 3);
#pragma unroll
    for (int n = 0; n < 8; ++n) {
      const int col = (wn * 8 + n) * 16 + (lane & 15);
      const float bv = bias[col];
      if (p == 2) {
        // h stored transposed: contiguous 16B store per lane.
        v8bf t0, t1;
#pragma unroll
        for (int v = 0; v < 8; ++v) {
          t0[v] = (bf16_t)(acc0[n][v] + bv);
          t1[v] = (bf16_t)(acc1[n][v] + bv);
        }
        *(v8bf*)(hT + col * PITCH_B + rb0) = t0;
        *(v8bf*)(hT + col * PITCH_B + rb1) = t1;
      } else {
        bf16_t* dst = (p == 0) ? fB : gB;
#pragma unroll
        for (int v = 0; v < 8; ++v) {
          dst[(rb0 + v) * PITCH_X + col] = (bf16_t)(acc0[n][v] + bv);
          dst[(rb1 + v) * PITCH_X + col] = (bf16_t)(acc1[n][v] + bv);
        }
      }
    }
  }
  __syncthreads();

  // ---- Phase 3: s = g * f^T -> f32 scores in reused xb space ----
  {
    const int m0 = wave * 16;
    v8f acc[8];
#pragma unroll
    for (int n = 0; n < 8; ++n) acc[n] = zero8();
#pragma clang loop unroll(disable)
    for (int k = 0; k < 8; ++k) {
      const int k0 = k * 32;
      v16bf a = load_frag_a(gB + m0 * PITCH_X + k0, PITCH_X);
#pragma unroll
      for (int n = 0; n < 8; ++n) {
        v16bf b = load_frag_b(fB + (n * 16) * PITCH_X + k0, PITCH_X);
        acc[n] = wmma_bf16(a, b, acc[n]);
      }
    }
    const int rb = m0 + ((lane >> 4) << 3);
#pragma unroll
    for (int n = 0; n < 8; ++n) {
      const int col = n * 16 + (lane & 15);
#pragma unroll
      for (int v = 0; v < 8; ++v) sS[(rb + v) * PITCH_S + col] = acc[n][v];
    }
  }
  __syncthreads();

  // ---- Softmax over each row of s (f32), write beta as bf16 into f space ---
  if (tid < SA_W) {
    const float* srow = sS + tid * PITCH_S;
    float m = -3.402823466e+38f;
    for (int j = 0; j < SA_W; ++j) m = fmaxf(m, srow[j]);
    float sum = 0.f;
    for (int j = 0; j < SA_W; ++j) sum += __expf(srow[j] - m);
    const float r = 1.f / sum;
    bf16_t* brow = beta + tid * PITCH_B;
    for (int j = 0; j < SA_W; ++j)
      brow[j] = (bf16_t)(__expf(srow[j] - m) * r);
  }
  __syncthreads();

  // ---- Phase 4: o = beta * h (K = 128), then out = o * x ----
  // Two N-halves of 8 accumulators each to bound register pressure.
  {
    const int m0 = wave * 16;
    const int rb = m0 + ((lane >> 4) << 3);
    const int cl = lane & 15;
#pragma clang loop unroll(disable)
    for (int half = 0; half < 2; ++half) {
      v8f acc[8];
#pragma unroll
      for (int n = 0; n < 8; ++n) acc[n] = zero8();
#pragma clang loop unroll(disable)
      for (int k = 0; k < 4; ++k) {
        const int k0 = k * 32;
        v16bf a = load_frag_a(beta + m0 * PITCH_B + k0, PITCH_B);
#pragma unroll
        for (int n = 0; n < 8; ++n) {
          const int c0 = (half * 8 + n) * 16;
          v16bf b = load_frag_b(hT + c0 * PITCH_B + k0, PITCH_B);
          acc[n] = wmma_bf16(a, b, acc[n]);
        }
      }
#pragma unroll
      for (int n = 0; n < 8; ++n) {
        const int col = (half * 8 + n) * 16 + cl;
#pragma unroll
        for (int v = 0; v < 8; ++v) {
          const int idx = (rb + v) * SA_C + col;
          orow[idx] = acc[n][v] * xrow[idx];
        }
      }
    }
  }
}

extern "C" void kernel_launch(void* const* d_in, const int* in_sizes, int n_in,
                              void* d_out, int out_size, void* d_ws, size_t ws_size,
                              hipStream_t stream) {
  (void)in_sizes; (void)n_in; (void)out_size; (void)ws_size;
  const float* x  = (const float*)d_in[0];
  const float* Wf = (const float*)d_in[1];
  const float* bf = (const float*)d_in[2];
  const float* Wg = (const float*)d_in[3];
  const float* bg = (const float*)d_in[4];
  const float* Wh = (const float*)d_in[5];
  const float* bh = (const float*)d_in[6];
  bf16_t* wt = (bf16_t*)d_ws;              // 3 * 256*256 bf16 = 384 KiB

  sa_prep_weights<<<(3 * SA_C * SA_C + 255) / 256, 256, 0, stream>>>(Wf, Wg, Wh, wt);

  hipFuncSetAttribute((const void*)sa_row_attention,
                      hipFuncAttributeMaxDynamicSharedMemorySize, SMEM_BYTES);
  sa_row_attention<<<16 * 128, 256, SMEM_BYTES, stream>>>(
      x, wt, bf, bg, bh, (float*)d_out);
}
